// SparseMoEMultiHeadAttention_51874615001516
// MI455X (gfx1250) — matrix operand order repaired and checked by reading, not compiled, not run on hardware
//
#include <hip/hip_runtime.h>
#include <hip/hip_bf16.h>
#include <math.h>

// ---------------- module hyperparameters ----------------
#define BB   2
#define LL   2048
#define DD   1024
#define EE   8
#define KK   2
#define HH   16
#define HDD  64
#define NKV  8            // H / K
#define KV   512          // NKV * HD
#define TT   (BB * LL)    // 4096 tokens

// ---------------- WMMA types / helpers ----------------
typedef __bf16 bf16;
typedef __attribute__((ext_vector_type(16))) __bf16 v16bf;
typedef __attribute__((ext_vector_type(8)))  __bf16 v8bf;
typedef __attribute__((ext_vector_type(8)))  float  v8f;
typedef __attribute__((ext_vector_type(4)))  unsigned int v4u;
typedef __attribute__((ext_vector_type(8)))  int v8i;
typedef __attribute__((ext_vector_type(4)))  int v4i;

union FragU { v16bf v; v8bf h[2]; };

// A operand (16x32, row-major source, ld in elements).
// Lane L: row = L&15; lanes 0-15 hold K 0..7 (V0-3) and 16..23 (V4-7);
// lanes 16-31 hold K 8..15 and 24..31.  (ISA 7.12.2, 16-bit A 16x32)
static __device__ __forceinline__ v16bf load_frag_a(const bf16* base, int ld) {
  const int lane = threadIdx.x & 31;
  const int row  = lane & 15;
  const int k0   = (lane < 16) ? 0 : 8;
  const bf16* p  = base + (size_t)row * ld + k0;
  FragU u;
  u.h[0] = *(const v8bf*)(p);        // K k0..k0+7
  u.h[1] = *(const v8bf*)(p + 16);   // K k0+16..k0+23
  return u.v;
}

// B operand (32x16) for C = A * B^T where B is stored row-major [N][K], ld over K.
// Lane L: col n = L&15; lanes 0-15 hold K 0..7 and 16..23, lanes 16-31 hold K 8..15 and 24..31.
static __device__ __forceinline__ v16bf load_frag_b(const bf16* base, int ld) {
  const int lane = threadIdx.x & 31;
  const int col  = lane & 15;
  const int k0   = (lane < 16) ? 0 : 8;
  const bf16* p  = base + (size_t)col * ld + k0;
  FragU u;
  u.h[0] = *(const v8bf*)(p);        // K k0..k0+7
  u.h[1] = *(const v8bf*)(p + 16);   // K k0+16..k0+23
  return u.v;
}

static __device__ __forceinline__ v8f wmma_bf16(v16bf a, v16bf b, v8f c) {
  return __builtin_amdgcn_wmma_f32_16x16x32_bf16(false, a, false, b, (short)0, c,
                                                 false, false);
}

// ---------------- TDM: DMA a [128 rows x 32 cols] bf16 tile into LDS ----------------
// D# per CDNA5 ISA.  2-D tile, data_size=2B,
// dim0 = contiguous K direction (32 elems), dim1 = 128 rows of stride ldb.
static __device__ __forceinline__ void tdm_load_b_tile(const bf16* gaddr,
                                                       unsigned lds_off,
                                                       unsigned ldb_elems) {
  const unsigned long long ga = (unsigned long long)(uintptr_t)gaddr;
  v4u g0;
  g0[0] = 1u;                                            // count=1, user mode
  g0[1] = lds_off;                                       // lds_addr (bytes)
  g0[2] = (unsigned)(ga & 0xffffffffu);                  // global_addr[31:0]
  g0[3] = (unsigned)((ga >> 32) & 0x01ffffffu)           // global_addr[56:32]
        | (2u << 30);                                    // type=2 ("image")
  v8i g1;
  g1[0] = (int)(1u << 16);          // workgroup_mask=0, data_size=1 (2 bytes)
  g1[1] = (int)(32u << 16);         // tensor_dim0[15:0]=32
  g1[2] = (int)(128u << 16);        // tensor_dim0[31:16]=0 | tensor_dim1[15:0]=128
  g1[3] = (int)(32u << 16);         // tensor_dim1[31:16]=0 | tile_dim0=32
  g1[4] = (int)128u;                // tile_dim1=128, tile_dim2=0 (2-D)
  g1[5] = (int)ldb_elems;           // tensor_dim0_stride[31:0] (data_size units)
  g1[6] = 0;                        // stride hi bits / dim1_stride (unused, 2-D)
  g1[7] = 0;
  v4i z4 = {0, 0, 0, 0};
#if __clang_major__ >= 23
  v8i z8 = {0, 0, 0, 0, 0, 0, 0, 0};
  __builtin_amdgcn_tensor_load_to_lds(g0, g1, z4, z4, z8, 0);
#else
  __builtin_amdgcn_tensor_load_to_lds(g0, g1, z4, z4, 0);
#endif
}

// ---------------- f32 -> bf16 conversion ----------------
__global__ void cvt_f32_bf16_kernel(const float* __restrict__ src,
                                    bf16* __restrict__ dst, size_t n) {
  size_t i = (size_t)blockIdx.x * blockDim.x + threadIdx.x;
  size_t stride = (size_t)gridDim.x * blockDim.x;
  for (; i < n; i += stride) dst[i] = (bf16)src[i];
}

// ---------------- router: noisy top-2 gating ----------------
__global__ __launch_bounds__(32)
void router_kernel(const float* __restrict__ x, const float* __restrict__ noise,
                   const float* __restrict__ Wr, const float* __restrict__ br,
                   const float* __restrict__ Wn, const float* __restrict__ bn,
                   int* __restrict__ topi, float* __restrict__ gates) {
  const int t = blockIdx.x;
  const int lane = threadIdx.x;
  const float* xp = x + (size_t)t * DD;
  float accr[EE], accn[EE];
  #pragma unroll
  for (int e = 0; e < EE; ++e) { accr[e] = 0.f; accn[e] = 0.f; }
  for (int d = lane; d < DD; d += 32) {
    float xv = xp[d];
    #pragma unroll
    for (int e = 0; e < EE; ++e) {
      accr[e] += xv * Wr[e * DD + d];
      accn[e] += xv * Wn[e * DD + d];
    }
  }
  #pragma unroll
  for (int e = 0; e < EE; ++e) {
    #pragma unroll
    for (int m = 16; m >= 1; m >>= 1) {
      accr[e] += __shfl_xor(accr[e], m, 32);
      accn[e] += __shfl_xor(accn[e], m, 32);
    }
  }
  if (lane == 0) {
    float noisy[EE];
    #pragma unroll
    for (int e = 0; e < EE; ++e) {
      float lg = accr[e] + br[e];
      float nl = accn[e] + bn[e];
      float sp = (nl > 20.f) ? nl : log1pf(expf(nl));   // softplus
      noisy[e] = lg + noise[(size_t)t * EE + e] * sp;
    }
    int i0 = 0;
    #pragma unroll
    for (int e = 1; e < EE; ++e) if (noisy[e] > noisy[i0]) i0 = e;
    int i1 = (i0 == 0) ? 1 : 0;
    #pragma unroll
    for (int e = 0; e < EE; ++e)
      if (e != i0 && noisy[e] > noisy[i1]) i1 = e;
    float v0 = noisy[i0], v1 = noisy[i1];
    float e0 = expf(v0 - v0), e1 = expf(v1 - v0);
    float s = e0 + e1;
    topi[t * KK + 0] = i0;  topi[t * KK + 1] = i1;
    gates[t * KK + 0] = e0 / s;  gates[t * KK + 1] = e1 / s;
  }
}

// ---------------- generic WMMA GEMM: C[M,N] = A[M,Kd] * B[N,Kd]^T ----------------
// 8 waves/block; wave (wm,wn) owns a 32x64 tile; block tile = 128x128.
// B k-slab (128x32) is DMA'd into LDS by the TDM once per k-step and shared by
// all 8 waves; A fragments stream from global (L2-resident, per-wave rows).
// OUT_MODE: 0 -> store bf16; 1 -> store f32 = acc + bias[n]; 2 -> f32 C += acc.
template <int OUT_MODE>
__global__ __launch_bounds__(256)
void gemm_abt_kernel(const bf16* __restrict__ A, int lda,
                     const bf16* __restrict__ Bm, int ldb,
                     void* __restrict__ Cv, int ldc,
                     const float* __restrict__ bias, int Kd) {
  __shared__ bf16 blB[128 * 32];   // [n_local][k] row-major, 8 KB

  const int lane = threadIdx.x & 31;
  const int w  = threadIdx.x >> 5;
  const int wm = w & 3, wn = w >> 2;
  const int m0 = blockIdx.y * 128 + wm * 32;
  const int n0blk = blockIdx.x * 128;
  const int nLoc = wn * 64;

  const unsigned ldsOffB = (unsigned)(uintptr_t)&blB[0];

  v8f acc[2][4];
  #pragma unroll
  for (int i = 0; i < 2; ++i)
    #pragma unroll
    for (int j = 0; j < 4; ++j)
      #pragma unroll
      for (int r = 0; r < 8; ++r) acc[i][j][r] = 0.f;

  for (int kb = 0; kb < Kd; kb += 32) {
    __syncthreads();                 // previous k-step done reading blB
    if (w == 0) {
      tdm_load_b_tile(Bm + (size_t)n0blk * ldb + kb, ldsOffB, (unsigned)ldb);
      __builtin_amdgcn_s_wait_tensorcnt(0);
    }
    __syncthreads();                 // tile landed, visible to all waves

    if (kb + 32 < Kd)                // hint next A rows (global_prefetch_b8)
      __builtin_prefetch(A + (size_t)m0 * lda + kb + 32, 0, 3);

    v16bf a0 = load_frag_a(A + (size_t)(m0)      * lda + kb, lda);
    v16bf a1 = load_frag_a(A + (size_t)(m0 + 16) * lda + kb, lda);
    #pragma unroll
    for (int j = 0; j < 4; ++j) {
      v16bf bf = load_frag_b(blB + (size_t)(nLoc + j * 16) * 32, 32);
      acc[0][j] = wmma_bf16(a0, bf, acc[0][j]);
      acc[1][j] = wmma_bf16(a1, bf, acc[1][j]);
    }
  }

  const int nIdx = lane & 15, hi = lane >> 4;
  #pragma unroll
  for (int i = 0; i < 2; ++i) {
    #pragma unroll
    for (int j = 0; j < 4; ++j) {
      #pragma unroll
      for (int r = 0; r < 8; ++r) {
        const int m = m0 + i * 16 + r + hi * 8;
        const int n = n0blk + nLoc + j * 16 + nIdx;
        const size_t idx = (size_t)m * ldc + n;
        if constexpr (OUT_MODE == 0) {
          ((bf16*)Cv)[idx] = (bf16)acc[i][j][r];
        } else if constexpr (OUT_MODE == 1) {
          ((float*)Cv)[idx] = acc[i][j][r] + bias[n];
        } else {
          ((float*)Cv)[idx] += acc[i][j][r];
        }
      }
    }
  }
}

// ---------------- gather top-K expert query slices ----------------
__global__ void gather_q_kernel(const bf16* __restrict__ allq,
                                const int* __restrict__ topi,
                                bf16* __restrict__ qb) {
  size_t idx = (size_t)blockIdx.x * blockDim.x + threadIdx.x;
  const size_t total = (size_t)TT * KK * KV;
  const size_t stride = (size_t)gridDim.x * blockDim.x;
  for (; idx < total; idx += stride) {
    size_t t = idx / (KK * KV);
    int rem  = (int)(idx % (KK * KV));
    int j = rem / KV, v = rem % KV;
    int e = topi[t * KK + j];
    qb[idx] = allq[t * (size_t)(EE * KV) + (size_t)e * KV + v];
  }
}

// ---------------- flash attention (causal, online softmax, WMMA) ----------------
// grid: (L/16, H, B); one wave handles 16 query rows of one head.
__global__ __launch_bounds__(32)
void flash_attn_kernel(const bf16* __restrict__ Qb, const bf16* __restrict__ Kb,
                       const bf16* __restrict__ Vb, bf16* __restrict__ AOb) {
  const int qt   = blockIdx.x;
  const int h    = blockIdx.y;
  const int b    = blockIdx.z;
  const int lane = threadIdx.x & 31;
  const int nIdx = lane & 15;
  const int hi   = lane >> 4;

  const int kvh = h & (NKV - 1);
  const int jj  = h >> 3;            // which of the K=2 gate slots

  const bf16* Qp = Qb + ((size_t)(b * LL + qt * 16)) * (HH * HDD) + h * HDD;
  const bf16* Kp = Kb + (size_t)b * LL * KV + kvh * HDD;
  const bf16* Vp = Vb + (size_t)b * LL * KV + kvh * HDD;

  const v16bf qa0 = load_frag_a(Qp, HH * HDD);        // hd 0..31
  const v16bf qa1 = load_frag_a(Qp + 32, HH * HDD);   // hd 32..63

  float mrow[8], lrow[8];
  v8f o[4];
  #pragma unroll
  for (int r = 0; r < 8; ++r) { mrow[r] = -__builtin_inff(); lrow[r] = 0.f; }
  #pragma unroll
  for (int t2 = 0; t2 < 4; ++t2)
    #pragma unroll
    for (int r = 0; r < 8; ++r) o[t2][r] = 0.f;

  __shared__ bf16 vl[32][64 + 8];   // V block  (stride 144B, 16B aligned)
  __shared__ bf16 pl[16][32 + 8];   // P block  (stride 80B, 16B aligned)

  const int kvEnd = qt * 16 + 16;
  for (int kv0 = 0; kv0 < kvEnd; kv0 += 32) {
    // stage V rows kv0..kv0+31 into LDS (extra rows are causal-masked -> P==0)
    {
      const bf16* vrow = Vp + (size_t)(kv0 + lane) * KV;
      #pragma unroll
      for (int c = 0; c < 64; c += 8)
        *(v8bf*)&vl[lane][c] = *(const v8bf*)(vrow + c);
    }

    // S = Q K^T : two 16x16 tiles, each with K-reduction over HD=64 (2 WMMA)
    v8f s[2];
    #pragma unroll
    for (int nt = 0; nt < 2; ++nt) {
      v8f a;
      #pragma unroll
      for (int r = 0; r < 8; ++r) a[r] = 0.f;
      v16bf kb0 = load_frag_b(Kp + (size_t)(kv0 + nt * 16) * KV + 0,  KV);
      v16bf kb1 = load_frag_b(Kp + (size_t)(kv0 + nt * 16) * KV + 32, KV);
      a = wmma_bf16(qa0, kb0, a);
      a = wmma_bf16(qa1, kb1, a);
      s[nt] = a;
    }

    // scale + causal mask (HD^-0.5 = 0.125)
    #pragma unroll
    for (int r = 0; r < 8; ++r) {
      const int qg = qt * 16 + r + hi * 8;
      #pragma unroll
      for (int nt = 0; nt < 2; ++nt) {
        const int kg = kv0 + nt * 16 + nIdx;
        float v = s[nt][r] * 0.125f;
        s[nt][r] = (kg <= qg) ? v : -__builtin_inff();
      }
    }

    // online softmax (row reductions across the 16 lanes of each half)
    #pragma unroll
    for (int r = 0; r < 8; ++r) {
      float tmax = fmaxf(s[0][r], s[1][r]);
      tmax = fmaxf(tmax, __shfl_xor(tmax, 1, 32));
      tmax = fmaxf(tmax, __shfl_xor(tmax, 2, 32));
      tmax = fmaxf(tmax, __shfl_xor(tmax, 4, 32));
      tmax = fmaxf(tmax, __shfl_xor(tmax, 8, 32));
      const float mnew  = fmaxf(mrow[r], tmax);
      const float scale = expf(mrow[r] - mnew);
      mrow[r] = mnew;
      const float p0 = expf(s[0][r] - mnew);
      const float p1 = expf(s[1][r] - mnew);
      float ps = p0 + p1;
      ps += __shfl_xor(ps, 1, 32);
      ps += __shfl_xor(ps, 2, 32);
      ps += __shfl_xor(ps, 4, 32);
      ps += __shfl_xor(ps, 8, 32);
      lrow[r] = lrow[r] * scale + ps;
      #pragma unroll
      for (int t2 = 0; t2 < 4; ++t2) o[t2][r] *= scale;
      const int row = r + hi * 8;
      pl[row][nIdx]      = (bf16)p0;   // C-layout -> LDS
      pl[row][16 + nIdx] = (bf16)p1;
    }

    // P (A operand via LDS, layout fix-up) x V (B operand from LDS, strided)
    const v16bf pa = load_frag_a(&pl[0][0], 32 + 8);
    #pragma unroll
    for (int t2 = 0; t2 < 4; ++t2) {
      const int col = t2 * 16 + nIdx;
      const int k0  = hi * 8;
      v16bf vb;
      #pragma unroll
      for (int i = 0; i < 16; ++i) vb[i] = vl[k0 + i + (i >= 8 ? 8 : 0)][col];
      o[t2] = wmma_bf16(pa, vb, o[t2]);
    }
  }

  // normalize + store attention output: AO[t, jj*KV + kvh*HD + hd] (bf16)
  bf16* outp = AOb + (size_t)(b * LL + qt * 16) * (KK * KV) + jj * KV + kvh * HDD;
  #pragma unroll
  for (int r = 0; r < 8; ++r) {
    const int m = r + hi * 8;
    const float inv = 1.0f / lrow[r];
    #pragma unroll
    for (int t2 = 0; t2 < 4; ++t2)
      outp[(size_t)m * (KK * KV) + t2 * 16 + nIdx] = (bf16)(o[t2][r] * inv);
  }
}

// ---------------- gated scatter: dense t[T,E,KV] = one_hot(topi)*gates*ao ----------------
__global__ void scatter_t_kernel(const bf16* __restrict__ aob,
                                 const int* __restrict__ topi,
                                 const float* __restrict__ gates,
                                 bf16* __restrict__ tb) {
  size_t idx = (size_t)blockIdx.x * blockDim.x + threadIdx.x;
  const size_t total = (size_t)TT * EE * KV;
  const size_t stride = (size_t)gridDim.x * blockDim.x;
  for (; idx < total; idx += stride) {
    size_t t = idx / (EE * KV);
    int rem  = (int)(idx % (EE * KV));
    int e = rem / KV, v = rem % KV;
    float a = 0.f;
    #pragma unroll
    for (int j = 0; j < KK; ++j)
      if (topi[t * KK + j] == e)
        a += gates[t * KK + j] * (float)aob[t * (size_t)(KK * KV) + (size_t)j * KV + v];
    tb[idx] = (bf16)a;
  }
}

// ---------------- host-side launch ----------------
extern "C" void kernel_launch(void* const* d_in, const int* in_sizes, int n_in,
                              void* d_out, int out_size, void* d_ws, size_t ws_size,
                              hipStream_t stream) {
  (void)in_sizes; (void)n_in; (void)out_size; (void)ws_size;
  const float* x     = (const float*)d_in[0];
  const float* noise = (const float*)d_in[1];
  const float* Wr    = (const float*)d_in[2];
  const float* br    = (const float*)d_in[3];
  const float* Wn    = (const float*)d_in[4];
  const float* bn    = (const float*)d_in[5];
  const float* Win   = (const float*)d_in[6];
  const float* Wout  = (const float*)d_in[7];
  const float* Wk    = (const float*)d_in[8];
  const float* Wv    = (const float*)d_in[9];
  const float* bias  = (const float*)d_in[10];
  float* out = (float*)d_out;

  // carve workspace (256B aligned slices)
  char* p = (char*)d_ws;
  auto carve = [&](size_t bytes) -> void* {
    void* r = (void*)p;
    p += (bytes + 255) & ~(size_t)255;
    return r;
  };
  bf16* xb    = (bf16*)carve((size_t)TT * DD * sizeof(bf16));
  bf16* winb  = (bf16*)carve((size_t)EE * KV * DD * sizeof(bf16));
  bf16* woutb = (bf16*)carve((size_t)EE * DD * KV * sizeof(bf16));
  bf16* wkb   = (bf16*)carve((size_t)KV * DD * sizeof(bf16));
  bf16* wvb   = (bf16*)carve((size_t)KV * DD * sizeof(bf16));
  bf16* allq  = (bf16*)carve((size_t)TT * EE * KV * sizeof(bf16));
  bf16* kbuf  = (bf16*)carve((size_t)TT * KV * sizeof(bf16));
  bf16* vbuf  = (bf16*)carve((size_t)TT * KV * sizeof(bf16));
  bf16* qbuf  = (bf16*)carve((size_t)TT * KK * KV * sizeof(bf16));
  bf16* aob   = (bf16*)carve((size_t)TT * KK * KV * sizeof(bf16));
  bf16* tbuf  = (bf16*)carve((size_t)TT * EE * KV * sizeof(bf16));
  int*  topi  = (int*)carve((size_t)TT * KK * sizeof(int));
  float* gat  = (float*)carve((size_t)TT * KK * sizeof(float));

  auto cvt = [&](const float* s, bf16* d, size_t n) {
    int blocks = (int)((n + 255) / 256);
    if (blocks > 2048) blocks = 2048;
    cvt_f32_bf16_kernel<<<blocks, 256, 0, stream>>>(s, d, n);
  };
  cvt(x,    xb,    (size_t)TT * DD);
  cvt(Win,  winb,  (size_t)EE * KV * DD);
  cvt(Wout, woutb, (size_t)EE * DD * KV);
  cvt(Wk,   wkb,   (size_t)KV * DD);
  cvt(Wv,   wvb,   (size_t)KV * DD);

  // router
  router_kernel<<<TT, 32, 0, stream>>>(x, noise, Wr, br, Wn, bn, topi, gat);

  // k/v projections: [T,D] x [KV,D]^T -> bf16 [T,KV]
  {
    dim3 grid(KV / 128, TT / 128);
    gemm_abt_kernel<0><<<grid, 256, 0, stream>>>(xb, DD, wkb, DD, (void*)kbuf, KV,
                                                 nullptr, DD);
    gemm_abt_kernel<0><<<grid, 256, 0, stream>>>(xb, DD, wvb, DD, (void*)vbuf, KV,
                                                 nullptr, DD);
  }

  // dense all-expert q projection (matches reference einsum), then gather top-K
  for (int e = 0; e < EE; ++e) {
    dim3 grid(KV / 128, TT / 128);
    gemm_abt_kernel<0><<<grid, 256, 0, stream>>>(
        xb, DD, winb + (size_t)e * KV * DD, DD,
        (void*)(allq + (size_t)e * KV), EE * KV, nullptr, DD);
  }
  gather_q_kernel<<<2048, 256, 0, stream>>>(allq, topi, qbuf);

  // causal flash attention
  {
    dim3 grid(LL / 16, HH, BB);
    flash_attn_kernel<<<grid, 32, 0, stream>>>(qbuf, kbuf, vbuf, aob);
  }

  // gated scatter to dense t, then 8 accumulating output GEMMs (+bias on first)
  scatter_t_kernel<<<2048, 256, 0, stream>>>(aob, topi, gat, tbuf);
  for (int e = 0; e < EE; ++e) {
    dim3 grid(DD / 128, TT / 128);
    if (e == 0)
      gemm_abt_kernel<1><<<grid, 256, 0, stream>>>(
          tbuf + (size_t)e * KV, EE * KV,
          woutb + (size_t)e * DD * KV, KV, (void*)out, DD, bias, KV);
    else
      gemm_abt_kernel<2><<<grid, 256, 0, stream>>>(
          tbuf + (size_t)e * KV, EE * KV,
          woutb + (size_t)e * DD * KV, KV, (void*)out, DD, bias, KV);
  }
}
